// Encoder_84499186581951
// MI455X (gfx1250) — matrix operand order repaired and checked by reading, not compiled, not run on hardware
//
#include <hip/hip_runtime.h>
#include <hip/hip_bf16.h>
#include <math.h>

#define B_   32
#define T_   2048
#define V_   128
#define H_   512
#define G4_  2048          // 4*H
#define NWG  32            // persistent workgroups
#define HS   16            // hidden units per WG
// per-WG gate columns = 4*HS = 64 ; Wh slice in LDS = 512*64 bf16 = 64KB

typedef __bf16 bf16_t;
typedef __attribute__((ext_vector_type(16))) __bf16 v16bf;
typedef __attribute__((ext_vector_type(8)))  __bf16 v8bf;
typedef __attribute__((ext_vector_type(8)))  float  v8f;

// ---------------- init: zero h state and grid-barrier counter ----------------
__global__ void k_init(bf16_t* __restrict__ h, unsigned* __restrict__ ctr) {
    int i = blockIdx.x * blockDim.x + threadIdx.x;
    if (i == 0) *ctr = 0u;
    if (i < B_ * H_) h[i] = (bf16_t)0.0f;
}

// ---------------- one-hot -> token index (argmax over V=128) ----------------
__global__ void k_tokens(const float* __restrict__ x, int* __restrict__ tok) {
    int i = blockIdx.x * blockDim.x + threadIdx.x;
    if (i >= B_ * T_) return;
    const float* row = x + (size_t)i * V_;
    int t = 0;
    #pragma unroll 4
    for (int j = 0; j < V_; ++j)
        if (row[j] > 0.5f) t = j;
    tok[i] = t;
}

// ---------------- per-batch output timestep = first(token==1) else T-1 ------
__global__ void k_outt(const int* __restrict__ tok, int* __restrict__ out_t) {
    int b = threadIdx.x;
    if (b >= B_) return;
    int r = T_ - 1;
    for (int t = 0; t < T_; ++t) {
        if (tok[b * T_ + t] == 1) { r = t; break; }
    }
    out_t[b] = r;
}

// ---------------- Wib = Wi + bh (fold bias into the gather table) -----------
__global__ void k_wib(const float* __restrict__ Wi, const float* __restrict__ bh,
                      float* __restrict__ Wib) {
    int i = blockIdx.x * blockDim.x + threadIdx.x;
    if (i < V_ * G4_) Wib[i] = Wi[i] + bh[i & (G4_ - 1)];
}

__device__ __forceinline__ float sigf(float x) {
    return 1.0f / (1.0f + __expf(-x));
}

// ---------------- persistent LSTM recurrence kernel -------------------------
// grid = 32 WGs x 256 threads (8 waves). WG w owns hidden units [16w, 16w+16).
__global__ __launch_bounds__(256, 1)
void k_lstm(const float* __restrict__ Wh,     // [512][2048] f32
            const float* __restrict__ Wib,    // [128][2048] f32 (Wi + bh)
            const int*   __restrict__ tok,    // [32][2048]
            const int*   __restrict__ out_t,  // [32]
            bf16_t*      __restrict__ h,      // [32][512] bf16 (global state)
            float*       __restrict__ out,    // [32][512] f32
            unsigned*    __restrict__ ctr)    // grid barrier counter
{
    extern __shared__ char smem[];
    bf16_t* ldsB = (bf16_t*)smem;                          // 64KB: B fragments
    float*  zbuf = (float*)(smem + 65536);                 //  8KB: z[4][32][16]
    float*  cst  = (float*)(smem + 65536 + 8192);          //  2KB: c[32][16]
    int*    otts = (int*)  (smem + 65536 + 8192 + 2048);   // 32 ints

    const int wg    = blockIdx.x;       // 0..31
    const int tid   = threadIdx.x;      // 0..255
    const int lane  = tid & 31;
    const int wv    = tid >> 5;         // 0..7
    const int mtile = wv & 1;           // batch tile: rows 0-15 / 16-31
    const int gate  = wv >> 1;          // 0=i 1=f 2=g 3=o
    const int hidb  = wg * HS;

    // ---- stage Wh slice into LDS, bf16, WMMA B-fragment order ----
    // element e -> { g:2 | kk:4 | lane:5 | i:4 } ; lane (n = l&15, khalf = l>>4)
// holds B[K = kk*32 + (i>>3)*16 + khalf*8 + (i&7)][N = n] for gate-column g*512+hidb+n.
    for (int e = tid; e < 4 * 16 * 32 * 16; e += 256) {
        int i  = e & 15;
        int l  = (e >> 4) & 31;
        int kk = (e >> 9) & 15;
        int g  = (e >> 13) & 3;
        int k  = kk * 32 + (i >> 3) * 16 + (l >> 4) * 8 + (i & 7);
        int col = g * H_ + hidb + (l & 15);
        ldsB[e] = (bf16_t)Wh[(size_t)k * G4_ + col];
    }
    for (int e = tid; e < B_ * HS; e += 256) cst[e] = 0.0f;  // c0 = 0
    if (tid < B_) otts[tid] = out_t[tid];
    __syncthreads();

    const bf16_t* ldsBw = ldsB + gate * (16 * 32 * 16);
    const int kh = lane >> 4;    // which K-half this lane covers in A/B frags
    const int ml = lane & 15;    // A row within tile / B-D column
    const int gcol = gate * H_ + hidb + ml;   // this lane's global gate column

    for (int t = 0; t < T_; ++t) {
        // ---- token gather for this lane's 8 D-rows (independent of WMMA:
        //      compiler can hoist these loads to overlap the GEMM chain) ----
        float xi[8];
        #pragma unroll
        for (int r = 0; r < 8; ++r) {
            int m = mtile * 16 + kh * 8 + r;
            xi[r] = Wib[(size_t)tok[m * T_ + t] * G4_ + gcol];
        }

        // ---- GEMM tile: acc(16x16) += h[mtile-rows] @ Wh[gate-cols], K=512
        //      two independent 8-deep accumulate chains for latency slack ----
        v8f acc0 = {}, acc1 = {};
        const bf16_t* hrow = h + (size_t)(mtile * 16 + ml) * H_ + kh * 8;
        #pragma unroll
        for (int kk = 0; kk < 8; ++kk) {
            v8bf alo = *(const v8bf*)(hrow + kk * 32);
            v8bf ahi = *(const v8bf*)(hrow + kk * 32 + 16);
            v16bf a;
            #pragma unroll
            for (int i = 0; i < 8; ++i) { a[i] = alo[i]; a[i + 8] = ahi[i]; }
            v16bf b = *(const v16bf*)(ldsBw + ((kk * 32 + lane) << 4));
            acc0 = __builtin_amdgcn_wmma_f32_16x16x32_bf16(
                       false, a, false, b, (short)0, acc0, false, false);

            v8bf clo = *(const v8bf*)(hrow + (kk + 8) * 32);
            v8bf chi = *(const v8bf*)(hrow + (kk + 8) * 32 + 16);
            v16bf a2;
            #pragma unroll
            for (int i = 0; i < 8; ++i) { a2[i] = clo[i]; a2[i + 8] = chi[i]; }
            v16bf b2 = *(const v16bf*)(ldsBw + (((kk + 8) * 32 + lane) << 4));
            acc1 = __builtin_amdgcn_wmma_f32_16x16x32_bf16(
                       false, a2, false, b2, (short)0, acc1, false, false);
        }

        // ---- z = acc + (Wi[token] + bh) ; stash tile in LDS ----
        #pragma unroll
        for (int r = 0; r < 8; ++r) {
            int m = mtile * 16 + kh * 8 + r;      // D layout: VGPR r, half kh
            float z = acc0[r] + acc1[r] + xi[r];
            zbuf[(gate * B_ + m) * HS + ml] = z;
        }
        __syncthreads();

        // ---- gate nonlinearities + state update (512 elems / 256 thr) ----
        for (int e = tid; e < B_ * HS; e += 256) {
            int m = e >> 4, nn = e & 15;
            float zi = zbuf[(0 * B_ + m) * HS + nn];
            float zf = zbuf[(1 * B_ + m) * HS + nn];
            float zg = zbuf[(2 * B_ + m) * HS + nn];
            float zo = zbuf[(3 * B_ + m) * HS + nn];
            float c  = sigf(zf) * cst[e] + sigf(zi) * tanhf(zg);
            cst[e] = c;
            float hv = sigf(zo) * tanhf(c);
            h[(size_t)m * H_ + hidb + nn] = (bf16_t)hv;
            if (t == otts[m]) out[(size_t)m * H_ + hidb + nn] = hv;
        }

        // ---- grid barrier (monotonic counter, no reset needed) ----
        __syncthreads();
        __threadfence();
        if (tid == 0) {
            atomicAdd(ctr, 1u);
            const unsigned target = (unsigned)NWG * (unsigned)(t + 1);
            while (__hip_atomic_load(ctr, __ATOMIC_ACQUIRE,
                                     __HIP_MEMORY_SCOPE_AGENT) < target) {
                __builtin_amdgcn_s_sleep(2);
            }
            __threadfence();
        }
        __syncthreads();
    }
}

// ---------------------------------------------------------------------------
extern "C" void kernel_launch(void* const* d_in, const int* in_sizes, int n_in,
                              void* d_out, int out_size, void* d_ws, size_t ws_size,
                              hipStream_t stream) {
    const float* inputs = (const float*)d_in[0];   // [32][2048][128]
    const float* Wi     = (const float*)d_in[1];   // [128][2048]
    const float* Wh     = (const float*)d_in[2];   // [512][2048]
    const float* bh     = (const float*)d_in[3];   // [2048]
    float* out = (float*)d_out;                    // [32][512]

    // workspace carve (16B aligned chunks)
    char* ws = (char*)d_ws;
    bf16_t*   h    = (bf16_t*)ws;                        size_t o = 32768;        // 32*512 bf16
    float*    Wib  = (float*)(ws + o);                   o += (size_t)V_*G4_*4;   // 1MB
    int*      tokb = (int*)(ws + o);                     o += (size_t)B_*T_*4;    // 256KB
    int*      outt = (int*)(ws + o);                     o += 128;
    unsigned* ctr  = (unsigned*)(ws + o);

    k_init  <<<(B_*H_ + 255)/256, 256, 0, stream>>>(h, ctr);
    k_tokens<<<(B_*T_ + 255)/256, 256, 0, stream>>>(inputs, tokb);
    k_outt  <<<1, 32, 0, stream>>>(tokb, outt);
    k_wib   <<<(V_*G4_ + 255)/256, 256, 0, stream>>>(Wi, bh, Wib);

    const size_t smem = 65536 + 8192 + 2048 + 256;   // B frags + z + c + otts
    k_lstm<<<NWG, 256, smem, stream>>>(Wh, Wib, tokb, outt, h, out, ctr);
}
